// Attention_44573170598190
// MI455X (gfx1250) — matrix operand order repaired and checked by reading, not compiled, not run on hardware
//
#include <hip/hip_runtime.h>
#include <hip/hip_bf16.h>

typedef __attribute__((ext_vector_type(16))) _Float16 v16h;
typedef __attribute__((ext_vector_type(8)))  _Float16 v8h;
typedef __attribute__((ext_vector_type(8)))  float    v8f;
typedef __attribute__((ext_vector_type(4)))  int      v4i;

#define NN   64
#define QN   512
#define KN   512
#define CIN  256
#define HID  256
#define NH   8
#define CH   32

#ifndef __has_builtin
#define __has_builtin(x) 0
#endif
#if __has_builtin(__builtin_amdgcn_global_load_async_to_lds_b128)
#define HAVE_ASYNC_LDS 1
#else
#define HAVE_ASYNC_LDS 0
#endif

// typed pointers in explicit address spaces for the async copy builtin
typedef __attribute__((address_space(1))) v4i gv4i;   // global
typedef __attribute__((address_space(3))) v4i lv4i;   // LDS

__device__ __forceinline__ v8f wmma16(v16h a, v16h b, v8f c) {
    return __builtin_amdgcn_wmma_f32_16x16x32_f16(false, a, false, b, (short)0, c, false, false);
}

__device__ __forceinline__ v16h cat8(v8h lo, v8h hi) {
    v16h r;
#pragma unroll
    for (int i = 0; i < 8; ++i) { r[i] = lo[i]; r[8 + i] = hi[i]; }
    return r;
}

__device__ __forceinline__ float halfwave_max(float v) {
    v = fmaxf(v, __shfl_xor(v, 1, 32));
    v = fmaxf(v, __shfl_xor(v, 2, 32));
    v = fmaxf(v, __shfl_xor(v, 4, 32));
    v = fmaxf(v, __shfl_xor(v, 8, 32));
    return v;
}
__device__ __forceinline__ float halfwave_sum(float v) {
    v += __shfl_xor(v, 1, 32);
    v += __shfl_xor(v, 2, 32);
    v += __shfl_xor(v, 4, 32);
    v += __shfl_xor(v, 8, 32);
    return v;
}

// ---------------------------------------------------------------------------
// K0: pack 5 f32 weight matrices [in=256][out=256] -> f16 transposed [out][in]
// order: wq, wk, wv, wg, wo
// ---------------------------------------------------------------------------
__global__ void pack_w_kernel(const float* __restrict__ wq, const float* __restrict__ wk,
                              const float* __restrict__ wv, const float* __restrict__ wg,
                              const float* __restrict__ wo, _Float16* __restrict__ wT) {
    int idx = blockIdx.x * 256 + threadIdx.x;
    if (idx >= 5 * 65536) return;
    int m = idx >> 16;
    int r = idx & 65535;
    int o = r >> 8;      // out col
    int i = r & 255;     // in row
    const float* w = (m == 0) ? wq : (m == 1) ? wk : (m == 2) ? wv : (m == 3) ? wg : wo;
    wT[(size_t)m * 65536 + (size_t)o * 256 + i] = (_Float16)w[(size_t)i * 256 + o];
}

// ---------------------------------------------------------------------------
// K1: projections. mode 0: q (scaled 1/sqrt(32)), 1: k, 2: v (transposed),
//     3: gate (sigmoid(x@wg + bg)).
// Each wave: 16 rows x 32 cols (head h == wave id). grid = (NQ/16, 4).
// ---------------------------------------------------------------------------
__global__ void proj_kernel(const float* __restrict__ qx, const float* __restrict__ kx,
                            const float* __restrict__ vx, const _Float16* __restrict__ wT,
                            const float* __restrict__ bg,
                            _Float16* __restrict__ q16, _Float16* __restrict__ k16,
                            _Float16* __restrict__ vT16, _Float16* __restrict__ g16) {
    const int mode    = blockIdx.y;
    const int rowbase = blockIdx.x * 16;
    const int lane    = threadIdx.x & 31;
    const int wave    = threadIdx.x >> 5;      // == head h
    const int colbase = wave * 32;
    const int half    = lane >> 4;
    const int m       = lane & 15;             // row within tile / col within tile

    const float* x = (mode == 1) ? kx : (mode == 2) ? vx : qx;
    const _Float16* w = wT + (size_t)mode * 65536;
    const int arow = rowbase + m;

    v8f c0 = {}; v8f c1 = {};
#pragma unroll
    for (int kt = 0; kt < 8; ++kt) {
        const int kb = kt * 32;
        // ---- A tile: rows = token rows, Kdim = input channel (f32 -> f16) ----
        const float* xr = x + (size_t)arow * CIN + kb + half * 8;
        float4 a0 = *(const float4*)(xr);
        float4 a1 = *(const float4*)(xr + 4);
        float4 a2 = *(const float4*)(xr + 16);
        float4 a3 = *(const float4*)(xr + 20);
        v16h A;
        A[0]=(_Float16)a0.x; A[1]=(_Float16)a0.y; A[2]=(_Float16)a0.z; A[3]=(_Float16)a0.w;
        A[4]=(_Float16)a1.x; A[5]=(_Float16)a1.y; A[6]=(_Float16)a1.z; A[7]=(_Float16)a1.w;
        A[8]=(_Float16)a2.x; A[9]=(_Float16)a2.y; A[10]=(_Float16)a2.z; A[11]=(_Float16)a2.w;
        A[12]=(_Float16)a3.x; A[13]=(_Float16)a3.y; A[14]=(_Float16)a3.z; A[15]=(_Float16)a3.w;
        // ---- B tiles: two 8-half runs (k = 8*half.., 16 + 8*half..) from transposed weights ----
        v16h B0 = cat8(*(const v8h*)(w + (size_t)(colbase + m) * CIN + kb + half * 8), *(const v8h*)(w + (size_t)(colbase + m) * CIN + kb + 16 + half * 8));
        v16h B1 = cat8(*(const v8h*)(w + (size_t)(colbase + 16 + m) * CIN + kb + half * 8), *(const v8h*)(w + (size_t)(colbase + 16 + m) * CIN + kb + 16 + half * 8));
        c0 = wmma16(A, B0, c0);
        c1 = wmma16(A, B1, c1);
    }

    // ---- epilogue: C layout row = rowbase + r + 8*half, col = colbase + m ----
#pragma unroll
    for (int r = 0; r < 8; ++r) {
        const int row = rowbase + r + 8 * half;     // global token row (n*512 + q/k)
        const int n   = row >> 9;
        const int tq  = row & 511;
        const int h   = wave;
        float v0 = c0[r], v1 = c1[r];
        if (mode == 0) {                            // q, scaled
            v0 *= 0.17677669529663687f; v1 *= 0.17677669529663687f;
            size_t base = (((size_t)n * NH + h) * QN + tq) * CH;
            q16[base + m]      = (_Float16)v0;
            q16[base + 16 + m] = (_Float16)v1;
        } else if (mode == 1) {                     // k
            size_t base = (((size_t)n * NH + h) * KN + tq) * CH;
            k16[base + m]      = (_Float16)v0;
            k16[base + 16 + m] = (_Float16)v1;
        } else if (mode == 2) {                     // v transposed: [n][h][c][k]
            size_t base = (((size_t)n * NH + h) * CH) * KN + tq;
            vT16[base + (size_t)(m) * KN]      = (_Float16)v0;
            vT16[base + (size_t)(16 + m) * KN] = (_Float16)v1;
        } else {                                    // gate
            float b0 = bg[colbase + m], b1 = bg[colbase + 16 + m];
            float s0 = 1.0f / (1.0f + __expf(-(v0 + b0)));
            float s1 = 1.0f / (1.0f + __expf(-(v1 + b1)));
            size_t base = (size_t)row * HID + colbase;
            g16[base + m]      = (_Float16)s0;
            g16[base + 16 + m] = (_Float16)s1;
        }
    }
}

// ---------------------------------------------------------------------------
// K2: fused attention (flash-style online softmax) + gating.
// grid = (N*H, Q/128); block = 256 (8 waves x 16 q-rows).
// K/V tiles double-buffered in LDS via async global->LDS copies (ASYNCcnt).
// ---------------------------------------------------------------------------

// Stage one 32x32 K tile + one 32x32 V tile (4 KB total) into LDS.
// 256 threads, 16 bytes each. tid<128: K part, tid>=128: V part.
__device__ __forceinline__ void stage_kv(const _Float16* __restrict__ kp,
                                         const _Float16* __restrict__ vp,
                                         _Float16* dst, int k0, int tid) {
    const int part = tid >> 7;            // 0 = K tile, 1 = V tile
    const int t    = tid & 127;
    const int row  = t >> 2;              // 0..31
    const int seg  = (t & 3) * 8;         // halves: 0,8,16,24
    const _Float16* g = part ? (vp + (size_t)row * KN + k0 + seg)     // vT[c][k]
                             : (kp + (size_t)(k0 + row) * CH + seg);  // k16[k][c]
    _Float16* l = dst + part * 1024 + row * 32 + seg;
#if HAVE_ASYNC_LDS
    __builtin_amdgcn_global_load_async_to_lds_b128((gv4i*)g, (lv4i*)l, 0, 0);
#else
    *(v8h*)l = *(const v8h*)g;
#endif
}

__device__ __forceinline__ void stage_wait() {
#if HAVE_ASYNC_LDS
#if __has_builtin(__builtin_amdgcn_s_wait_asynccnt)
    __builtin_amdgcn_s_wait_asynccnt(0);
#else
    asm volatile("s_wait_asynccnt 0" ::: "memory");
#endif
#endif
    __syncthreads();
}

__global__ void attn_kernel(const _Float16* __restrict__ q16, const _Float16* __restrict__ k16,
                            const _Float16* __restrict__ vT16,
                            const float* __restrict__ bias_mask, const float* __restrict__ bias_pair,
                            const _Float16* __restrict__ g16, _Float16* __restrict__ og) {
    __shared__ _Float16 ldsP[8 * 16 * 32];          // per-wave 16x32 P staging (8 KB)
    __shared__ _Float16 kvS[2][2048];               // double-buffered K+V tiles (8 KB)

    const int nh   = blockIdx.x;
    const int n    = nh >> 3;
    const int h    = nh & 7;
    const int tid  = threadIdx.x;
    const int wave = tid >> 5;
    const int lane = tid & 31;
    const int half = lane >> 4;
    const int m    = lane & 15;
    const int qbase = blockIdx.y * 128 + wave * 16;

    const _Float16* kp = k16  + (size_t)nh * KN * CH;
    const _Float16* vp = vT16 + (size_t)nh * CH * KN;

    // ---- Q A-tile (register resident for whole K loop) ----
    const _Float16* qp = q16 + ((size_t)nh * QN + qbase) * CH;
    v16h QA;
    {
        const _Float16* r0 = qp + (size_t)m * CH + half * 16;
        QA = cat8(*(const v8h*)r0, *(const v8h*)(r0 + 8));
    }

    float mi[8], li[8];
#pragma unroll
    for (int r = 0; r < 8; ++r) { mi[r] = -1e30f; li[r] = 0.0f; }
    v8f acc0 = {}; v8f acc1 = {};

    _Float16* myP = ldsP + wave * (16 * 32);
    const float* bmrow = bias_mask + (size_t)n * KN;

    // prologue: stage first K/V tile pair
    stage_kv(kp, vp, &kvS[0][0], 0, tid);
    stage_wait();

    int buf = 0;
    for (int kt2 = 0; kt2 < 16; ++kt2) {
        const int k0 = kt2 * 32;
        // async-prefetch next tile pair into the other buffer
        if (kt2 + 1 < 16) stage_kv(kp, vp, &kvS[buf ^ 1][0], k0 + 32, tid);

        const _Float16* kb = &kvS[buf][0];
        const _Float16* vb = &kvS[buf][1024];

        // ---- K B-tiles from LDS (contiguous per lane) ----
        v16h KB0 = *(const v16h*)(kb + (m) * 32 + half * 16);
        v16h KB1 = *(const v16h*)(kb + (16 + m) * 32 + half * 16);
        v8f zero = {};
        v8f s0 = wmma16(QA, KB0, zero);
        v8f s1 = wmma16(QA, KB1, zero);

        const float bm0 = bmrow[k0 + m];
        const float bm1 = bmrow[k0 + 16 + m];

#pragma unroll
        for (int r = 0; r < 8; ++r) {
            const int qrow = qbase + r + 8 * half;
            const float* bp = bias_pair + ((size_t)h * QN + qrow) * KN + k0 + m;
            float v0 = s0[r] + bm0 + bp[0];
            float v1 = s1[r] + bm1 + bp[16];
            // online softmax over this 32-wide slab
            float t    = halfwave_max(fmaxf(v0, v1));
            float mnew = fmaxf(mi[r], t);
            float scl  = __expf(mi[r] - mnew);
            float p0   = __expf(v0 - mnew);
            float p1   = __expf(v1 - mnew);
            float rs   = halfwave_sum(p0 + p1);
            li[r] = li[r] * scl + rs;
            mi[r] = mnew;
            acc0[r] *= scl;
            acc1[r] *= scl;
            const int lr = r + 8 * half;            // C-layout row within tile
            myP[lr * 32 + m]      = (_Float16)p0;
            myP[lr * 32 + 16 + m] = (_Float16)p1;
        }
        asm volatile("s_wait_dscnt 0" ::: "memory");

        // ---- P A-tile from LDS (C-layout -> A-layout transit) ----
        v16h PA;
        {
            const _Float16* r0 = myP + m * 32 + half * 16;
            PA = cat8(*(const v8h*)r0, *(const v8h*)(r0 + 8));
        }
        // ---- V B-tiles from LDS ----
        v16h VB0 = *(const v16h*)(vb + m * 32 + half * 16);
        v16h VB1 = *(const v16h*)(vb + (16 + m) * 32 + half * 16);
        acc0 = wmma16(PA, VB0, acc0);
        acc1 = wmma16(PA, VB1, acc1);

        // next-tile async copies must land + all waves done with this buffer
        stage_wait();
        buf ^= 1;
    }

    // ---- epilogue: normalize, gate, store f16 [n][q][h*32+c] ----
#pragma unroll
    for (int r = 0; r < 8; ++r) {
        const int qrow = qbase + r + 8 * half;
        const float inv = 1.0f / li[r];
        float o0 = acc0[r] * inv;
        float o1 = acc1[r] * inv;
        size_t gi = ((size_t)n * QN + qrow) * HID + h * CH;
        o0 *= (float)g16[gi + m];
        o1 *= (float)g16[gi + 16 + m];
        og[gi + m]      = (_Float16)o0;
        og[gi + 16 + m] = (_Float16)o1;
    }
}

// ---------------------------------------------------------------------------
// K3: output projection out = og @ wo + bo  (f32 out). grid = NQ/16.
// ---------------------------------------------------------------------------
__global__ void outproj_kernel(const _Float16* __restrict__ og, const _Float16* __restrict__ woT,
                               const float* __restrict__ bo, float* __restrict__ out) {
    const int rowbase = blockIdx.x * 16;
    const int lane    = threadIdx.x & 31;
    const int wave    = threadIdx.x >> 5;
    const int colbase = wave * 32;
    const int half    = lane >> 4;
    const int m       = lane & 15;

    v8f c0 = {}; v8f c1 = {};
#pragma unroll
    for (int kt = 0; kt < 8; ++kt) {
        const int kb = kt * 32;
        const _Float16* ar = og + (size_t)(rowbase + m) * HID + kb;
        v16h A = cat8(*(const v8h*)(ar + half * 16), *(const v8h*)(ar + half * 16 + 8));
        v16h B0 = *(const v16h*)(woT + (size_t)(colbase + m) * HID + kb + half * 16);
        v16h B1 = *(const v16h*)(woT + (size_t)(colbase + 16 + m) * HID + kb + half * 16);
        c0 = wmma16(A, B0, c0);
        c1 = wmma16(A, B1, c1);
    }
#pragma unroll
    for (int r = 0; r < 8; ++r) {
        const int row = rowbase + r + 8 * half;
        out[(size_t)row * CIN + colbase + m]      = c0[r] + bo[colbase + m];
        out[(size_t)row * CIN + colbase + 16 + m] = c1[r] + bo[colbase + 16 + m];
    }
}

// ---------------------------------------------------------------------------
extern "C" void kernel_launch(void* const* d_in, const int* in_sizes, int n_in,
                              void* d_out, int out_size, void* d_ws, size_t ws_size,
                              hipStream_t stream) {
    const float* qx        = (const float*)d_in[0];
    const float* kx        = (const float*)d_in[1];
    const float* vx        = (const float*)d_in[2];
    const float* bias_mask = (const float*)d_in[3];
    const float* bias_pair = (const float*)d_in[4];
    const float* wq        = (const float*)d_in[5];
    const float* wk        = (const float*)d_in[6];
    const float* wv        = (const float*)d_in[7];
    const float* wg        = (const float*)d_in[8];
    const float* bg        = (const float*)d_in[9];
    const float* wo        = (const float*)d_in[10];
    const float* bo        = (const float*)d_in[11];
    float* out             = (float*)d_out;

    // workspace carve-up (f16 staging)
    char* ws = (char*)d_ws;
    const size_t PROJ_HALVES = (size_t)NN * NH * QN * CH;   // 8.39M halves each
    _Float16* wT   = (_Float16*)ws;                          ws += 5 * 65536 * sizeof(_Float16);
    _Float16* q16  = (_Float16*)ws;                          ws += PROJ_HALVES * sizeof(_Float16);
    _Float16* k16  = (_Float16*)ws;                          ws += PROJ_HALVES * sizeof(_Float16);
    _Float16* vT16 = (_Float16*)ws;                          ws += PROJ_HALVES * sizeof(_Float16);
    _Float16* g16  = (_Float16*)ws;                          ws += (size_t)NN * QN * HID * sizeof(_Float16);
    _Float16* og   = (_Float16*)ws;

    const int NQ = NN * QN;                                  // 32768 token rows

    pack_w_kernel<<<(5 * 65536) / 256, 256, 0, stream>>>(wq, wk, wv, wg, wo, wT);

    proj_kernel<<<dim3(NQ / 16, 4), 256, 0, stream>>>(qx, kx, vx, wT, bg,
                                                      q16, k16, vT16, g16);

    attn_kernel<<<dim3(NN * NH, QN / 128), 256, 0, stream>>>(q16, k16, vT16,
                                                             bias_mask, bias_pair, g16, og);

    outproj_kernel<<<NQ / 16, 256, 0, stream>>>(og, wT + 4 * 65536, bo, out);
}
